// Attention_90658169684243
// MI455X (gfx1250) — matrix operand order repaired and checked by reading, not compiled, not run on hardware
//
#include <hip/hip_runtime.h>
#include <hip/hip_bf16.h>

// ---------------------------------------------------------------------------
// Types / WMMA helpers (CDNA5 gfx1250, wave32)
// ---------------------------------------------------------------------------
typedef __attribute__((ext_vector_type(16))) __bf16 v16bf;
typedef __attribute__((ext_vector_type(8)))  __bf16 v8bf;
typedef __attribute__((ext_vector_type(8)))  float  v8f;

#define WMMA_BF16(a, b, c) \
  __builtin_amdgcn_wmma_f32_16x16x32_bf16(false, (a), false, (b), (short)0, (c), false, false)

#define LDS_STRIDE 40   // bf16 elements per LDS row (80 B, 16B aligned, bank-spread)

__device__ __forceinline__ float sigf(float x) { return 1.f / (1.f + expf(-x)); }

// Async global->LDS 16B copy (CDNA5 GLOBAL_LOAD_ASYNC_TO_LDS_B128, ASYNCcnt).
// LDS destination address = low 32 bits of the generic pointer (aperture rule).
__device__ __forceinline__ void async_copy_b128(void* lds_ptr, const void* gptr) {
  unsigned lds_addr = (unsigned)(uintptr_t)lds_ptr;
  unsigned long long ga = (unsigned long long)(uintptr_t)gptr;
  asm volatile("global_load_async_to_lds_b128 %0, %1, off"
               :: "v"(lds_addr), "v"(ga) : "memory");
}
__device__ __forceinline__ void async_wait0() {
  asm volatile("s_wait_asynccnt 0x0" ::: "memory");
}

// A fragment from row-major LDS tile As[m][k] (stride LDS_STRIDE).
// lanes 0-15: M=0..15, K = {0..7, 16..23}; lanes 16-31: K = {8..15, 24..31}.
__device__ __forceinline__ v16bf load_a_frag(const __bf16* As, int row0, int lane) {
  int m = row0 + (lane & 15);
  int khalf = (lane >> 4) << 3;
  const __bf16* p = As + m * LDS_STRIDE + khalf;
  v8bf lo = *(const v8bf*)(p);
  v8bf hi = *(const v8bf*)(p + 16);
  return __builtin_shufflevector(lo, hi, 0,1,2,3,4,5,6,7,8,9,10,11,12,13,14,15);
}

// B fragment from n-major LDS tile Bs[n][k] (stride LDS_STRIDE).
// lanes 0-15: N=lane, K = {0..7, 16..23}; lanes 16-31: N=lane-16, K = {8..15, 24..31}.
__device__ __forceinline__ v16bf load_b_frag(const __bf16* Bs, int nbase, int lane) {
  int n = nbase + (lane & 15);
  int kb = (lane >> 4) << 3;
  const __bf16* p = Bs + n * LDS_STRIDE + kb;
  v8bf lo = *(const v8bf*)(p);
  v8bf hi = *(const v8bf*)(p + 16);
  return __builtin_shufflevector(lo, hi, 0,1,2,3,4,5,6,7,8,9,10,11,12,13,14,15);
}

union CV4 { __bf16 h[4]; uint2 u; };

// ---------------------------------------------------------------------------
// Setup kernels
// ---------------------------------------------------------------------------

// conv_m2h_w (co, c, 3, 3) f32 -> wconv[co][tap*512 + c] bf16 (n-major, K=4608)
__global__ void k_convw(const float* __restrict__ src, __bf16* __restrict__ dst) {
  int idx = blockIdx.x * 256 + threadIdx.x;
  if (idx >= 512 * 4608) return;
  int co = idx / 4608;
  int r = idx - co * 4608;
  int tap = r >> 9, c = r & 511;
  dst[idx] = (__bf16)src[(size_t)(co * 512 + c) * 9 + tap];
}

// feature_map NCHW f32 -> fmx[(b*8+h)*32+w][c] bf16 (channel-last)
__global__ void k_fmconv(const float* __restrict__ src, __bf16* __restrict__ dst) {
  int idx = blockIdx.x * 256 + threadIdx.x;   // 33,554,432 total
  int c = idx & 511;
  int pos = idx >> 9;
  int w = pos & 31;
  int t2 = pos >> 5;
  int h = t2 & 7;
  int b = t2 >> 3;
  dst[idx] = (__bf16)src[(((size_t)b * 512 + c) * 8 + h) * 32 + w];
}

// W (N x K) f32 -> (N x Kpad) bf16, zero cols K..Kpad-1 (keeps n-major layout)
__global__ void k_wt(const float* __restrict__ src, __bf16* __restrict__ dst,
                     int N, int K, int Kpad) {
  int idx = blockIdx.x * 256 + threadIdx.x;
  if (idx >= N * Kpad) return;
  int n = idx / Kpad, k = idx - n * Kpad;
  dst[idx] = (k < K) ? (__bf16)src[(size_t)n * K + k] : (__bf16)0.0f;
}

__global__ void k_bsum(const float* a1, const float* b1, const float* a2, const float* b2,
                       float* s1, float* s2) {
  int i = blockIdx.x * 256 + threadIdx.x;
  if (i >= 2048) return;
  s1[i] = a1[i] + b1[i];
  s2[i] = a2[i] + b2[i];
}

__global__ void k_bhmean(const float* __restrict__ bh, float* __restrict__ out) {
  int idx = blockIdx.x * 256 + threadIdx.x;   // 131072
  int b = idx >> 9, k = idx & 511;
  float s = 0.f;
#pragma unroll 2
  for (int t = 0; t < 26; ++t) s += bh[((size_t)b * 26 + t) * 512 + k];
  out[idx] = s * (1.0f / 26.0f);
}

__global__ void k_h0c0(const float* hh, const float* hc,
                       float* h1, float* c1, float* h2, float* c2) {
  int idx = blockIdx.x * 256 + threadIdx.x;   // 131072
  float h0 = 0.5f * (hh[idx] + hh[131072 + idx]);
  float c0 = 0.5f * (hc[idx] + hc[131072 + idx]);
  h1[idx] = h0; h2[idx] = h0;
  c1[idx] = c0; c2[idx] = c0;
}

// ---------------------------------------------------------------------------
// Conv 3x3 as implicit GEMM (WMMA bf16). M=65536 (b,h,w), N=512, K=4608.
// WG 256 thr (8 waves): 64x128 tile; wave = 32x32 (4 WMMA / chunk).
// Double-buffered async global->LDS staging (copies overlap WMMA).
// ---------------------------------------------------------------------------
__global__ void k_conv_gemm(const __bf16* __restrict__ fmx, const __bf16* __restrict__ wconv,
                            const float* __restrict__ bias, __bf16* __restrict__ fmh) {
  __shared__ __align__(16) __bf16 As[2][64][LDS_STRIDE];
  __shared__ __align__(16) __bf16 Bs[2][128][LDS_STRIDE];
  int tid = threadIdx.x, lane = tid & 31, wave = tid >> 5;
  int wm = wave & 1, wn = wave >> 1;
  int m0 = blockIdx.x * 64;
  int n0 = blockIdx.y * 128;
  v8f acc00 = {}, acc01 = {}, acc10 = {}, acc11 = {};
  // Per-thread fixed task coordinates
  int ar = tid >> 2, ak8 = (tid & 3) << 3;          // A: 64 rows x 4 x (8 bf16)
  int am = m0 + ar;
  int ab = am >> 8, ahw = am & 255;
  int bn = tid >> 1, bkh = (tid & 1) << 4;          // B: 128 n x 2 x (16 bf16)

  auto issue = [&](int kc, int buf) {
    int tap = kc >> 4;
    int dh = tap / 3 - 1, dw = tap % 3 - 1;
    int h = (ahw >> 5) + dh, w = (ahw & 31) + dw;
    if ((unsigned)h < 8u && (unsigned)w < 32u) {
      async_copy_b128(&As[buf][ar][ak8],
                      fmx + (size_t)(((ab << 3) + h) * 32 + w) * 512 + ((kc & 15) << 5) + ak8);
    } else {
      uint4 z = {0u, 0u, 0u, 0u};
      *reinterpret_cast<uint4*>(&As[buf][ar][ak8]) = z;
    }
    const __bf16* src = wconv + (size_t)(n0 + bn) * 4608 + (kc << 5) + bkh;
    async_copy_b128(&Bs[buf][bn][bkh], src);
    async_copy_b128(&Bs[buf][bn][bkh + 8], src + 8);
  };

  issue(0, 0);
  for (int kc = 0; kc < 144; ++kc) {
    int buf = kc & 1;
    async_wait0();
    __syncthreads();
    if (kc + 1 < 144) issue(kc + 1, buf ^ 1);
    v16bf a0 = load_a_frag(&As[buf][0][0], wm * 32,      lane);
    v16bf a1 = load_a_frag(&As[buf][0][0], wm * 32 + 16, lane);
    v16bf b0 = load_b_frag(&Bs[buf][0][0], wn * 32,      lane);
    v16bf b1 = load_b_frag(&Bs[buf][0][0], wn * 32 + 16, lane);
    acc00 = WMMA_BF16(a0, b0, acc00);
    acc01 = WMMA_BF16(a0, b1, acc01);
    acc10 = WMMA_BF16(a1, b0, acc10);
    acc11 = WMMA_BF16(a1, b1, acc11);
  }
  int mlane = ((lane >> 4) << 3);
  int nc = n0 + wn * 32 + (lane & 15);
#pragma unroll
  for (int r = 0; r < 8; ++r) {
    int ma = m0 + wm * 32 + mlane + r;
    int mb = ma + 16;
    fmh[(size_t)ma * 512 + nc]      = (__bf16)(acc00[r] + bias[nc]);
    fmh[(size_t)ma * 512 + nc + 16] = (__bf16)(acc01[r] + bias[nc + 16]);
    fmh[(size_t)mb * 512 + nc]      = (__bf16)(acc10[r] + bias[nc]);
    fmh[(size_t)mb * 512 + nc + 16] = (__bf16)(acc11[r] + bias[nc + 16]);
  }
}

// ---------------------------------------------------------------------------
// Generic dual-input GEMM: OUT(MxN,f32) = A1(f32 MxK1) @ W1(bf16 NxK1).T
//                                       [+ A2 @ W2.T] [+ bias[n]] [+ addend[m,n]]
// K multiples of 32 (weights pre-padded). grid=(M/64, N/128).
// Double-buffered: B tile async global->LDS; A tile f32->bf16 VGPR convert.
// ---------------------------------------------------------------------------
__device__ __forceinline__ void gemm_pass(const float* __restrict__ A,
                                          const __bf16* __restrict__ W, int K, int N,
                                          int m0, int n0, int tid, int lane, int wm, int wn,
                                          __bf16 (*As)[64][LDS_STRIDE],
                                          __bf16 (*Bs)[128][LDS_STRIDE],
                                          v8f& acc00, v8f& acc01, v8f& acc10, v8f& acc11) {
  auto issue = [&](int k0, int buf) {
    {   // B tile: two 16B async copies per thread
      int n = tid >> 1, kh = (tid & 1) << 4;
      const __bf16* src = W + (size_t)(n0 + n) * K + k0 + kh;
      async_copy_b128(&Bs[buf][n][kh], src);
      async_copy_b128(&Bs[buf][n][kh + 8], src + 8);
    }
#pragma unroll
    for (int i = 0; i < 2; ++i) {   // A tile: 64 rows x 32 k f32 -> bf16
      int g = tid + i * 256;
      int r = g >> 3, k4 = (g & 7) << 2;
      float4 f = *reinterpret_cast<const float4*>(A + (size_t)(m0 + r) * K + k0 + k4);
      CV4 cv;
      cv.h[0] = (__bf16)f.x; cv.h[1] = (__bf16)f.y;
      cv.h[2] = (__bf16)f.z; cv.h[3] = (__bf16)f.w;
      *reinterpret_cast<uint2*>(&As[buf][r][k4]) = cv.u;
    }
  };
  __syncthreads();          // protect buffers across pass boundary
  issue(0, 0);
  int nk = K >> 5;
  for (int ck = 0; ck < nk; ++ck) {
    int buf = ck & 1;
    async_wait0();
    __syncthreads();
    if (ck + 1 < nk) issue((ck + 1) << 5, buf ^ 1);
    v16bf a0 = load_a_frag(&As[buf][0][0], wm * 32,      lane);
    v16bf a1 = load_a_frag(&As[buf][0][0], wm * 32 + 16, lane);
    v16bf b0 = load_b_frag(&Bs[buf][0][0], wn * 32,      lane);
    v16bf b1 = load_b_frag(&Bs[buf][0][0], wn * 32 + 16, lane);
    acc00 = WMMA_BF16(a0, b0, acc00);
    acc01 = WMMA_BF16(a0, b1, acc01);
    acc10 = WMMA_BF16(a1, b0, acc10);
    acc11 = WMMA_BF16(a1, b1, acc11);
  }
}

__global__ void k_gemm_dual(const float* __restrict__ A1, const __bf16* __restrict__ W1, int K1,
                            const float* __restrict__ A2, const __bf16* __restrict__ W2, int K2,
                            const float* __restrict__ bias, const float* __restrict__ addend,
                            float* __restrict__ OUT, int N) {
  __shared__ __align__(16) __bf16 As[2][64][LDS_STRIDE];
  __shared__ __align__(16) __bf16 Bs[2][128][LDS_STRIDE];
  int tid = threadIdx.x, lane = tid & 31, wave = tid >> 5;
  int wm = wave & 1, wn = wave >> 1;
  int m0 = blockIdx.x * 64;
  int n0 = blockIdx.y * 128;
  v8f acc00 = {}, acc01 = {}, acc10 = {}, acc11 = {};
  gemm_pass(A1, W1, K1, N, m0, n0, tid, lane, wm, wn, As, Bs, acc00, acc01, acc10, acc11);
  if (A2)
    gemm_pass(A2, W2, K2, N, m0, n0, tid, lane, wm, wn, As, Bs, acc00, acc01, acc10, acc11);
  int mlane = ((lane >> 4) << 3);
  int nc = n0 + wn * 32 + (lane & 15);
#pragma unroll
  for (int r = 0; r < 8; ++r) {
    int ma = m0 + wm * 32 + mlane + r;
    int mb = ma + 16;
    float v00 = acc00[r], v01 = acc01[r], v10 = acc10[r], v11 = acc11[r];
    if (bias) {
      float b0 = bias[nc], b1 = bias[nc + 16];
      v00 += b0; v01 += b1; v10 += b0; v11 += b1;
    }
    if (addend) {
      v00 += addend[(size_t)ma * N + nc];
      v01 += addend[(size_t)ma * N + nc + 16];
      v10 += addend[(size_t)mb * N + nc];
      v11 += addend[(size_t)mb * N + nc + 16];
    }
    OUT[(size_t)ma * N + nc]      = v00;
    OUT[(size_t)ma * N + nc + 16] = v01;
    OUT[(size_t)mb * N + nc]      = v10;
    OUT[(size_t)mb * N + nc + 16] = v11;
  }
}

// ---------------------------------------------------------------------------
// Attention kernels (fmh bf16 channel-last, 64MB -> L2-resident sweeps)
// ---------------------------------------------------------------------------
__global__ void k_attn_e(const __bf16* __restrict__ fmh, const float* __restrict__ q,
                         const float* __restrict__ wsc, const float* __restrict__ sb,
                         float* __restrict__ e) {
  int tid = threadIdx.x, lane = tid & 31, wave = tid >> 5;
  int gw = blockIdx.x * 8 + wave;   // 65536 waves total
  int b = gw >> 8, hw = gw & 255;
  const __bf16* fp = fmh + ((size_t)b * 256 + hw) * 512;
  const float* qp = q + (size_t)b * 512;
  float acc = 0.f;
#pragma unroll 4
  for (int c = lane; c < 512; c += 32)
    acc += tanhf((float)fp[c] + qp[c]) * wsc[c];
#pragma unroll
  for (int s = 16; s > 0; s >>= 1) acc += __shfl_xor(acc, s, 32);
  if (lane == 0) e[(size_t)b * 256 + hw] = acc + sb[0];
}

__global__ void k_softmax(const float* __restrict__ e, float* __restrict__ alpha) {
  __shared__ float red[256];
  int b = blockIdx.x, i = threadIdx.x;
  float x = e[(size_t)b * 256 + i];
  red[i] = x; __syncthreads();
  for (int s = 128; s > 0; s >>= 1) { if (i < s) red[i] = fmaxf(red[i], red[i + s]); __syncthreads(); }
  float m = red[0]; __syncthreads();
  float ex = expf(x - m);
  red[i] = ex; __syncthreads();
  for (int s = 128; s > 0; s >>= 1) { if (i < s) red[i] += red[i + s]; __syncthreads(); }
  alpha[(size_t)b * 256 + i] = ex / red[0];
}

__global__ void k_context(const __bf16* __restrict__ fmh, const float* __restrict__ alpha,
                          float* __restrict__ ctx) {
  __shared__ float al[256];
  int b = blockIdx.x, c = threadIdx.x;   // 512 threads
  if (c < 256) al[c] = alpha[(size_t)b * 256 + c];
  __syncthreads();
  const __bf16* fp = fmh + (size_t)b * 256 * 512 + c;
  float acc = 0.f;
#pragma unroll 4
  for (int hw = 0; hw < 256; ++hw) acc += al[hw] * (float)fp[(size_t)hw * 512];
  ctx[(size_t)b * 512 + c] = acc;
}

__global__ void k_xbuild(const float* __restrict__ ctx, const int* __restrict__ text, int t,
                         float* __restrict__ x) {
  int idx = blockIdx.x * 256 + threadIdx.x;   // 147456
  int b = idx / 576, j = idx - b * 576;
  float v = 0.f;
  if (j < 512) v = ctx[(size_t)b * 512 + j];
  else if (j < 550) v = (text[b * 26 + t] == (j - 512)) ? 1.f : 0.f;
  x[idx] = v;
}

__global__ void k_lstm(const float* __restrict__ g, const float* __restrict__ c_in,
                       float* __restrict__ h_out, float* __restrict__ c_out,
                       float* __restrict__ outh, int t) {
  int idx = blockIdx.x * 256 + threadIdx.x;   // 131072
  int b = idx >> 9, k = idx & 511;
  const float* gb = g + (size_t)b * 2048;
  float gi = gb[k], gf = gb[512 + k], gg = gb[1024 + k], go = gb[1536 + k];
  float c2 = sigf(gf) * c_in[idx] + sigf(gi) * tanhf(gg);
  float h2 = sigf(go) * tanhf(c2);
  h_out[idx] = h2;
  c_out[idx] = c2;
  if (outh) outh[((size_t)b * 26 + t) * 512 + k] = h2;
}

__global__ void k_gen(const float* __restrict__ outh, const float* __restrict__ gw,
                      const float* __restrict__ gb, float* __restrict__ out) {
  int idx = blockIdx.x * 256 + threadIdx.x;
  if (idx >= 256 * 26 * 38) return;
  int n = idx % 38;
  int bt = idx / 38;
  const float* hp = outh + (size_t)bt * 512;
  const float* wp = gw + (size_t)n * 512;
  float acc = gb[n];
#pragma unroll 4
  for (int k = 0; k < 512; ++k) acc += hp[k] * wp[k];
  out[idx] = acc;
}

// ---------------------------------------------------------------------------
// Host
// ---------------------------------------------------------------------------
extern "C" void kernel_launch(void* const* d_in, const int* in_sizes, int n_in,
                              void* d_out, int out_size, void* d_ws, size_t ws_size,
                              hipStream_t stream) {
  (void)in_sizes; (void)n_in; (void)out_size; (void)ws_size;
  const float* feature_map = (const float*)d_in[0];
  const float* batch_H     = (const float*)d_in[1];
  const float* hidden_h    = (const float*)d_in[2];
  const float* hidden_c    = (const float*)d_in[3];
  const int*   text        = (const int*)  d_in[4];
  const float* i2h_w       = (const float*)d_in[5];
  const float* h2h_w       = (const float*)d_in[6];
  const float* h2h_b       = (const float*)d_in[7];
  const float* conv_m2h_w  = (const float*)d_in[8];
  const float* conv_m2h_b  = (const float*)d_in[9];
  const float* conv_h2h_w  = (const float*)d_in[10];
  const float* conv_h2h_b  = (const float*)d_in[11];
  const float* score_w     = (const float*)d_in[12];
  const float* score_b     = (const float*)d_in[13];
  const float* rnn1_w_ih   = (const float*)d_in[14];
  const float* rnn1_w_hh   = (const float*)d_in[15];
  const float* rnn1_b_ih   = (const float*)d_in[16];
  const float* rnn1_b_hh   = (const float*)d_in[17];
  const float* hlin_w      = (const float*)d_in[18];
  const float* hlin_b      = (const float*)d_in[19];
  const float* rnn2_w_ih   = (const float*)d_in[20];
  const float* rnn2_w_hh   = (const float*)d_in[21];
  const float* rnn2_b_ih   = (const float*)d_in[22];
  const float* rnn2_b_hh   = (const float*)d_in[23];
  const float* gen_w       = (const float*)d_in[24];
  const float* gen_b       = (const float*)d_in[25];
  float* out = (float*)d_out;

  size_t off = 0;
  char* base = (char*)d_ws;
  auto alloc = [&](size_t bytes) -> void* {
    void* p = base + off;
    off = (off + bytes + 255) & ~(size_t)255;
    return p;
  };
  __bf16* fmx     = (__bf16*)alloc((size_t)33554432 * 2);
  __bf16* fmh     = (__bf16*)alloc((size_t)33554432 * 2);
  __bf16* wconv   = (__bf16*)alloc((size_t)512 * 4608 * 2);
  __bf16* wt_i2h  = (__bf16*)alloc((size_t)262144 * 2);
  __bf16* wt_h2h  = (__bf16*)alloc((size_t)262144 * 2);
  __bf16* wt_1x1  = (__bf16*)alloc((size_t)262144 * 2);
  __bf16* wt_hlin = (__bf16*)alloc((size_t)262144 * 2);
  __bf16* wt_r1ih = (__bf16*)alloc((size_t)2048 * 576 * 2);
  __bf16* wt_r1hh = (__bf16*)alloc((size_t)2048 * 512 * 2);
  __bf16* wt_r2ih = (__bf16*)alloc((size_t)2048 * 512 * 2);
  __bf16* wt_r2hh = (__bf16*)alloc((size_t)2048 * 512 * 2);
  float* bsum1   = (float*)alloc(2048 * 4);
  float* bsum2   = (float*)alloc(2048 * 4);
  float* bh_mean = (float*)alloc(131072 * 4);
  float* bh_proj = (float*)alloc(131072 * 4);
  float* h1b[2], *c1b[2], *h2b[2], *c2b[2];
  for (int i = 0; i < 2; ++i) {
    h1b[i] = (float*)alloc(131072 * 4);
    c1b[i] = (float*)alloc(131072 * 4);
    h2b[i] = (float*)alloc(131072 * 4);
    c2b[i] = (float*)alloc(131072 * 4);
  }
  float* vbuf = (float*)alloc(131072 * 4);
  float* qbuf = (float*)alloc(131072 * 4);
  float* ebuf = (float*)alloc(65536 * 4);
  float* abuf = (float*)alloc(65536 * 4);
  float* ctx  = (float*)alloc(131072 * 4);
  float* xbuf = (float*)alloc(147456 * 4);
  float* g1   = (float*)alloc(524288 * 4);
  float* g2   = (float*)alloc(524288 * 4);
  float* curb = (float*)alloc(131072 * 4);
  float* outh = (float*)alloc((size_t)3407872 * 4);

  dim3 blk(256);
  // ---- setup ----
  k_convw <<<dim3(9216),   blk, 0, stream>>>(conv_m2h_w, wconv);
  k_fmconv<<<dim3(131072), blk, 0, stream>>>(feature_map, fmx);
  k_wt<<<dim3(1024), blk, 0, stream>>>(i2h_w,      wt_i2h,  512, 512, 512);
  k_wt<<<dim3(1024), blk, 0, stream>>>(h2h_w,      wt_h2h,  512, 512, 512);
  k_wt<<<dim3(1024), blk, 0, stream>>>(conv_h2h_w, wt_1x1,  512, 512, 512);
  k_wt<<<dim3(1024), blk, 0, stream>>>(hlin_w,     wt_hlin, 512, 512, 512);
  k_wt<<<dim3(4608), blk, 0, stream>>>(rnn1_w_ih,  wt_r1ih, 2048, 550, 576);
  k_wt<<<dim3(4096), blk, 0, stream>>>(rnn1_w_hh,  wt_r1hh, 2048, 512, 512);
  k_wt<<<dim3(4096), blk, 0, stream>>>(rnn2_w_ih,  wt_r2ih, 2048, 512, 512);
  k_wt<<<dim3(4096), blk, 0, stream>>>(rnn2_w_hh,  wt_r2hh, 2048, 512, 512);
  k_bsum<<<dim3(8), blk, 0, stream>>>(rnn1_b_ih, rnn1_b_hh, rnn2_b_ih, rnn2_b_hh, bsum1, bsum2);
  k_bhmean<<<dim3(512), blk, 0, stream>>>(batch_H, bh_mean);
  k_h0c0<<<dim3(512), blk, 0, stream>>>(hidden_h, hidden_c, h1b[0], c1b[0], h2b[0], c2b[0]);

  // fmh = conv3x3(feature_map) + bias   (WMMA implicit GEMM, bf16 out)
  k_conv_gemm<<<dim3(1024, 4), blk, 0, stream>>>(fmx, wconv, conv_m2h_b, fmh);
  // bh_proj = mean_T(batch_H) @ i2h_w.T
  k_gemm_dual<<<dim3(4, 4), blk, 0, stream>>>(bh_mean, wt_i2h, 512,
                                              nullptr, nullptr, 0,
                                              nullptr, nullptr, bh_proj, 512);

  // ---- scan over T=26 ----
  int cur = 0;
  for (int t = 0; t < 26; ++t) {
    int nxt = cur ^ 1;
    k_gemm_dual<<<dim3(4, 4), blk, 0, stream>>>(h2b[cur], wt_h2h, 512,
                                                nullptr, nullptr, 0,
                                                h2h_b, bh_proj, vbuf, 512);
    k_gemm_dual<<<dim3(4, 4), blk, 0, stream>>>(vbuf, wt_1x1, 512,
                                                nullptr, nullptr, 0,
                                                conv_h2h_b, nullptr, qbuf, 512);
    k_attn_e <<<dim3(8192), blk, 0, stream>>>(fmh, qbuf, score_w, score_b, ebuf);
    k_softmax<<<dim3(256),  blk, 0, stream>>>(ebuf, abuf);
    k_context<<<dim3(256), dim3(512), 0, stream>>>(fmh, abuf, ctx);
    k_xbuild<<<dim3(576), blk, 0, stream>>>(ctx, text, t, xbuf);
    k_gemm_dual<<<dim3(4, 16), blk, 0, stream>>>(xbuf, wt_r1ih, 576,
                                                 h1b[cur], wt_r1hh, 512,
                                                 bsum1, nullptr, g1, 2048);
    k_lstm<<<dim3(512), blk, 0, stream>>>(g1, c1b[cur], h1b[nxt], c1b[nxt], nullptr, t);
    k_gemm_dual<<<dim3(4, 4), blk, 0, stream>>>(h1b[nxt], wt_hlin, 512,
                                                nullptr, nullptr, 0,
                                                hlin_b, nullptr, curb, 512);
    k_gemm_dual<<<dim3(4, 16), blk, 0, stream>>>(curb, wt_r2ih, 512,
                                                 h2b[cur], wt_r2hh, 512,
                                                 bsum2, nullptr, g2, 2048);
    k_lstm<<<dim3(512), blk, 0, stream>>>(g2, c2b[cur], h2b[nxt], c2b[nxt], outh, t);
    cur = nxt;
  }

  k_gen<<<dim3(988), blk, 0, stream>>>(outh, gen_w, gen_b, out);
}
